// iResBlock_63247688400972
// MI455X (gfx1250) — matrix operand order repaired and checked by reading, not compiled, not run on hardware
//
#include <hip/hip_runtime.h>
#include <cmath>

#define BB  65536
#define NN  128
#define HH  512
#define NPS 6

typedef __bf16 bf16;
typedef bf16  v16bf __attribute__((ext_vector_type(16)));
typedef float v8f   __attribute__((ext_vector_type(8)));

__device__ __forceinline__ v8f wmma_bf16(v16bf a, v16bf b, v8f c) {
  // (neg_a, A, neg_b, B, c_mod, C, reuse_a, reuse_b)
  return __builtin_amdgcn_wmma_f32_16x16x32_bf16(false, a, false, b, (short)0, c,
                                                 false, false);
}

// ---------------------------------------------------------------------------
// Fragment-resident storage. A WMMA operand fragment = 32 lanes x 16 bf16.
// Every matrix (weights in global, activations in LDS) is stored so that a
// lane's 16 bf16 are contiguous (32 B) -> one 32-byte vector load per lane,
// coalesced across the wave (1 KB contiguous per fragment).
// ---------------------------------------------------------------------------

// A-matrix [16 x K] (K multiple of 32): chunk kc covers columns kc*32..+31.
// Per ISA 7.12.2 (16-bit A 16x32): lane = row + 16*hi, element e holds
//   k_local = e<8 ? e + 8*hi : 16 + (e-8) + 8*hi.
// Inverse map (element position for matrix coord (m, n)):
__device__ __forceinline__ int afrag_idx(int m, int n) {
  const int kc  = n >> 5;
  const int kn  = n & 31;
  const int hif = (kn >> 3) & 1;
  const int e   = (kn & 7) | ((kn >> 4) << 3);
  return (((kc << 5) + m + (hif << 4)) << 4) + e;
}

__device__ __forceinline__ v16bf load_a_frag(const bf16* __restrict__ buf, int kc) {
  const int lane = threadIdx.x & 31;
  return *(const v16bf*)(buf + (((kc << 5) + lane) << 4));
}

// B-matrix [K x 16*NT] stored as tiles tile = nt*KC + kc (KC = K/32):
// lane = col + 16*hi, element e holds k_local = e<8 ? e + 8*hi : 16 + (e-8) + 8*hi.
__device__ __forceinline__ v16bf load_b_frag(const bf16* __restrict__ W, int tile) {
  const int lane = threadIdx.x & 31;
  return *(const v16bf*)(W + ((size_t)(tile << 5) + lane) * 16);
}

// Position of weight element (k, n) inside the fragment-packed array.
__host__ __device__ __forceinline__ size_t bfrag_idx(int k, int n, int KC) {
  const size_t tile = (size_t)(n >> 4) * KC + (k >> 5);
  const int lane = (n & 15) + (((k >> 3) & 1) << 4);
  return ((tile << 5) + lane) * 16 + ((k & 7) | (((k >> 4) & 1) << 3));
}

// CDNA5 has V_TANH_F32 (listed as a TRANS op in the S_DELAY_ALU rules).
__device__ __forceinline__ float fast_tanh(float a) {
#if __has_builtin(__builtin_amdgcn_tanhf)
  return __builtin_amdgcn_tanhf(a);
#elif __has_builtin(__builtin_amdgcn_rcpf)
  return 1.0f - 2.0f * __builtin_amdgcn_rcpf(__expf(2.0f * a) + 1.0f);
#else
  return 1.0f - 2.0f / (__expf(2.0f * a) + 1.0f);
#endif
}

// ---------------------------------------------------------------------------
// Prep: fp32 weights -> bf16 fragment-packed (W1, W2, W1^T, W2^T), lam, 1/lam.
// ---------------------------------------------------------------------------
__global__ void ires_prep_kernel(const float* __restrict__ W1,
                                 const float* __restrict__ W2,
                                 const float* __restrict__ Lambda,
                                 bf16* __restrict__ W1f,  // B of GEMM1: K=128
                                 bf16* __restrict__ W2f,  // B of GEMM2: K=512
                                 bf16* __restrict__ W1Tf, // B of VJP-2: K=512
                                 bf16* __restrict__ W2Tf, // B of VJP-1: K=128
                                 float* __restrict__ lam,
                                 float* __restrict__ rlam) {
  const int idx = blockIdx.x * blockDim.x + threadIdx.x;
  if (idx < NN * HH) {
    const int r = idx / HH;  // 0..127 (N)
    const int c = idx % HH;  // 0..511 (H)
    const float w1 = W1[idx];                 // W1[r][c], [N,H]
    W1f[bfrag_idx(r, c, NN / 32)]  = (bf16)w1;
    W1Tf[bfrag_idx(c, r, HH / 32)] = (bf16)w1;  // W1^T[c][r]
    const float w2 = W2[(size_t)c * NN + r];  // W2[c][r], [H,N]
    W2f[bfrag_idx(c, r, HH / 32)]  = (bf16)w2;
    W2Tf[bfrag_idx(r, c, NN / 32)] = (bf16)w2;  // W2^T[r][c]
  }
  if (idx < NN) {
    const float l = expf(Lambda[idx]);
    lam[idx] = l;
    rlam[idx] = 1.0f / l;
  }
}

// ---------------------------------------------------------------------------
// Main: one wave32 per 16-row tile.
// ---------------------------------------------------------------------------
__global__ __launch_bounds__(32)
void ires_tile_kernel(const float* __restrict__ x, const float* __restrict__ mask,
                      const float* __restrict__ veps, const float* __restrict__ mu,
                      const float* __restrict__ b1, const float* __restrict__ b2,
                      const bf16* __restrict__ W1f, const bf16* __restrict__ W2f,
                      const bf16* __restrict__ W1Tf, const bf16* __restrict__ W2Tf,
                      const float* __restrict__ lam, const float* __restrict__ rlam,
                      float* __restrict__ out, float* __restrict__ logdet) {
  __shared__ __align__(32) bf16  sh_h[16 * HH];   // tanh acts, A-frag order (16 KB)
  __shared__ __align__(32) bf16  sh_u[16 * HH];   // U' frag order          (16 KB)
  __shared__ __align__(32) bf16  sh_v[16 * NN];   // V / x_inp frag order   (4 KB)
  __shared__ float sh_ne[16 * NN];                // Neumann acc, linear    (8 KB)
  __shared__ float sh_red[16];

  const int lane = threadIdx.x & 31;
  const int col  = lane & 15;
  const int hi   = lane >> 4;
  const int m0   = blockIdx.x * 16;

  // ---- x_inp = (x - mu) * lam -> sh_v (fragment order) ----
  for (int idx = lane; idx < 16 * NN; idx += 32) {
    const int r = idx >> 7, c = idx & (NN - 1);
    const float xi = (x[(size_t)(m0 + r) * NN + c] - mu[c]) * lam[c];
    sh_v[afrag_idx(r, c)] = (bf16)xi;
  }
  __syncthreads();

  // ---- GEMM1: h = tanh(x_inp @ W1 + b1) ----
  {
    v16bf a4[4];
#pragma unroll
    for (int kc = 0; kc < 4; ++kc) a4[kc] = load_a_frag(sh_v, kc);
#pragma unroll 1
    for (int nt = 0; nt < HH / 16; ++nt) {
      v8f acc = {};
#pragma unroll
      for (int kc = 0; kc < 4; ++kc)
        acc = wmma_bf16(a4[kc], load_b_frag(W1f, nt * 4 + kc), acc);
#pragma unroll
      for (int v = 0; v < 8; ++v) {
        const int m = v + 8 * hi;
        const int n = nt * 16 + col;
        sh_h[afrag_idx(m, n)] = (bf16)fast_tanh(acc[v] + b1[n]);
      }
    }
  }
  __syncthreads();

  // ---- GEMM2: f_x = h @ W2 + b2; out = x - mu - (f_x/lam)*mask ----
#pragma unroll 1
  for (int nt = 0; nt < NN / 16; ++nt) {
    v8f acc = {};
#pragma unroll 4
    for (int kc = 0; kc < HH / 32; ++kc)
      acc = wmma_bf16(load_a_frag(sh_h, kc), load_b_frag(W2f, nt * 16 + kc), acc);
#pragma unroll
    for (int v = 0; v < 8; ++v) {
      const int m = v + 8 * hi;
      const int n = nt * 16 + col;
      const size_t gi = (size_t)(m0 + m) * NN + n;
      const float fx = acc[v] + b2[n];
      out[gi] = x[gi] - mu[n] - fx * rlam[n] * mask[gi];
    }
  }

  // ---- init V = vareps (frag order), neumann = vareps (linear fp32) ----
  for (int idx = lane; idx < 16 * NN; idx += 32) {
    const int r = idx >> 7, c = idx & (NN - 1);
    const float e = veps[(size_t)(m0 + r) * NN + c];
    sh_v[afrag_idx(r, c)] = (bf16)e;
    sh_ne[idx] = e;
  }
  __syncthreads();

  // ---- Neumann series (6 VJPs) + final VJP of the series sum ----
#pragma unroll 1
  for (int it = 0; it <= NPS; ++it) {
    if (it == NPS) {  // final pass: V := neumann (stop-gradient copy)
      for (int idx = lane; idx < 16 * NN; idx += 32) {
        const int r = idx >> 7, c = idx & (NN - 1);
        sh_v[afrag_idx(r, c)] = (bf16)sh_ne[idx];
      }
      __syncthreads();
    }
    // U' = (V @ W2^T) * (1 - h^2)
    {
      v16bf av[4];
#pragma unroll
      for (int kc = 0; kc < 4; ++kc) av[kc] = load_a_frag(sh_v, kc);
#pragma unroll 1
      for (int nt = 0; nt < HH / 16; ++nt) {
        v8f acc = {};
#pragma unroll
        for (int kc = 0; kc < 4; ++kc)
          acc = wmma_bf16(av[kc], load_b_frag(W2Tf, nt * 4 + kc), acc);
#pragma unroll
        for (int v = 0; v < 8; ++v) {
          const int m = v + 8 * hi;
          const int n = nt * 16 + col;
          const int fi = afrag_idx(m, n);
          const float hh = (float)sh_h[fi];
          sh_u[fi] = (bf16)(acc[v] * (1.0f - hh * hh));
        }
      }
    }
    __syncthreads();

    if (it < NPS) {
      // Vnew = U' @ W1^T; neumann -= Vnew; V = Vnew
#pragma unroll 1
      for (int nt = 0; nt < NN / 16; ++nt) {
        v8f acc = {};
#pragma unroll 4
        for (int kc = 0; kc < HH / 32; ++kc)
          acc = wmma_bf16(load_a_frag(sh_u, kc), load_b_frag(W1Tf, nt * 16 + kc), acc);
#pragma unroll
        for (int v = 0; v < 8; ++v) {
          const int m = v + 8 * hi;
          const int n = nt * 16 + col;
          sh_ne[m * NN + n] -= acc[v];
          sh_v[afrag_idx(m, n)] = (bf16)acc[v];
        }
      }
      __syncthreads();
    } else {
      // vjp_jac = U' @ W1^T; logdet[row] = sum_n vjp_jac * eps
      if (lane < 16) sh_red[lane] = 0.0f;
      __syncthreads();
      float part[8];
#pragma unroll
      for (int v = 0; v < 8; ++v) part[v] = 0.0f;
#pragma unroll 1
      for (int nt = 0; nt < NN / 16; ++nt) {
        v8f acc = {};
#pragma unroll 4
        for (int kc = 0; kc < HH / 32; ++kc)
          acc = wmma_bf16(load_a_frag(sh_u, kc), load_b_frag(W1Tf, nt * 16 + kc), acc);
#pragma unroll
        for (int v = 0; v < 8; ++v) {
          const int m = v + 8 * hi;
          const int n = nt * 16 + col;
          part[v] += acc[v] * veps[(size_t)(m0 + m) * NN + n];
        }
      }
#pragma unroll
      for (int v = 0; v < 8; ++v) atomicAdd(&sh_red[v + 8 * hi], part[v]);
      __syncthreads();
      if (lane < 16) logdet[m0 + lane] = sh_red[lane];
    }
  }
}

extern "C" void kernel_launch(void* const* d_in, const int* in_sizes, int n_in,
                              void* d_out, int out_size, void* d_ws, size_t ws_size,
                              hipStream_t stream) {
  const float* x    = (const float*)d_in[0];
  const float* mask = (const float*)d_in[1];
  const float* veps = (const float*)d_in[2];
  const float* Lam  = (const float*)d_in[3];
  const float* mu   = (const float*)d_in[4];
  const float* W1   = (const float*)d_in[5];
  const float* b1   = (const float*)d_in[6];
  const float* W2   = (const float*)d_in[7];
  const float* b2   = (const float*)d_in[8];

  float* out    = (float*)d_out;          // [B,N]
  float* logdet = out + (size_t)BB * NN;  // [B]

  // Workspace: fragment-packed bf16 weights (~512 KB, L2-resident) + lam/rlam.
  char* ws   = (char*)d_ws;
  bf16* W1f  = (bf16*)(ws + 0);
  bf16* W2f  = (bf16*)(ws + 131072);
  bf16* W1Tf = (bf16*)(ws + 262144);
  bf16* W2Tf = (bf16*)(ws + 393216);
  float* lam  = (float*)(ws + 524288);
  float* rlam = (float*)(ws + 524800);

  ires_prep_kernel<<<(NN * HH + 255) / 256, 256, 0, stream>>>(
      W1, W2, Lam, W1f, W2f, W1Tf, W2Tf, lam, rlam);

  ires_tile_kernel<<<BB / 16, 32, 0, stream>>>(
      x, mask, veps, mu, b1, b2, W1f, W2f, W1Tf, W2Tf, lam, rlam, out, logdet);
}
